// SimplifiedNet_69647189672594
// MI455X (gfx1250) — matrix operand order repaired and checked by reading, not compiled, not run on hardware
//
#include <hip/hip_runtime.h>

// CVAE-LSTM forward for gfx1250 (MI455X): f16 WMMA everywhere, persistent
// batch-partitioned LSTM kernels, generic WMMA GEMM for MLPs + logits.
// Logits stores are non-temporal (786MB streaming write must not rinse L2;
// the 192MB L2 should keep weights/activations hot instead).
//
// d_in leaf order (insertion-order flatten of setup_inputs()):
//  0 x (T*B i32), 1 condition (B i32), 2 eps (B*Z f32),
//  3 enc_emb, 4 dec_emb, 5 cvae_emb,
//  6..21  enc_lstm l0..l3 x (wih,whh,bih,bhh)
// 22..37  dec_lstm l0..l3 x (wih,whh,bih,bhh)
// 38 dec_fc.w, 39 dec_fc.b
// 40..47 cvae_enc (w,b)x4 ; 48..55 cvae_prior ; 56..63 cvae_dec

typedef _Float16 half_t;
typedef __attribute__((ext_vector_type(16))) _Float16 v16h;
typedef __attribute__((ext_vector_type(8)))  _Float16 v8h;
typedef __attribute__((ext_vector_type(8)))  float    v8f;

#define T_N 96
#define B_N 128
#define V_N 16000
#define E_N 64
#define H_N 128
#define Z_N 64
#define G_N 512   // 4*H

__device__ __forceinline__ float sigf(float x) { return 1.0f / (1.0f + __expf(-x)); }

__device__ __forceinline__ v8f zero8() {
  v8f z = {0.f, 0.f, 0.f, 0.f, 0.f, 0.f, 0.f, 0.f};
  return z;
}

// A fragment (16x32 f16, M x K): lanes 0-15 hold row M=lane, K = k0+[0..15];
// lanes 16-31 hold K = k0+16+[0..15] (contiguous, matching load_b_frag).
__device__ __forceinline__ v16h load_a_frag(const half_t* A, int ld, int m0, int k0, int lane) {
  const int r  = lane & 15;
  const int hs = (lane >> 4) << 4;            // 0 or 16
  const half_t* p = A + (size_t)(m0 + r) * ld + (k0 + hs);
  v8h lo = *(const v8h*)(p);
  v8h hi = *(const v8h*)(p + 8);
  v16h o;
#pragma unroll
  for (int i = 0; i < 8; ++i) { o[i] = lo[i]; o[i + 8] = hi[i]; }
  return o;
}

// B fragment (32x16 f16, K x N) loaded from row-major W (N_total x K):
// lane = output column, 16 contiguous K halves (lanes 0-15: K=k0..k0+15,
// lanes 16-31: K=k0+16..k0+31).
__device__ __forceinline__ v16h load_b_frag(const half_t* W, int ld, int n0, int k0, int lane) {
  const int c  = lane & 15;
  const int kb = (lane >> 4) << 4;            // 0 or 16
  return *(const v16h*)(W + (size_t)(n0 + c) * ld + (k0 + kb));
}

__device__ __forceinline__ v8f wmma16(v16h a, v16h b, v8f c) {
  return __builtin_amdgcn_wmma_f32_16x16x32_f16(false, a, false, b, (short)0, c, false, false);
}

// ---------------------------------------------------------------- utilities
__global__ void k_f2h(const float* __restrict__ s, half_t* __restrict__ d, int n) {
  int i = blockIdx.x * 256 + threadIdx.x;
  if (i < n) d[i] = (half_t)s[i];
}

// Embedding gather with row-0 zeroing; dec_mode forces token 1 for t==0.
__global__ void k_embed(const int* __restrict__ idx, const float* __restrict__ table,
                        half_t* __restrict__ out, int dec_mode) {
  int tpos = blockIdx.x;
  int e = threadIdx.x;
  int tok = idx[tpos];
  if (dec_mode && tpos < B_N) tok = 1;
  float v = (tok == 0) ? 0.0f : table[(size_t)tok * E_N + e];
  out[(size_t)tpos * E_N + e] = (half_t)v;
}

__global__ void k_build_encin(const float* __restrict__ h, const float* __restrict__ c,
                              const half_t* __restrict__ cond, half_t* __restrict__ out) {
  int i = blockIdx.x * 256 + threadIdx.x;
  if (i >= B_N * 1088) return;
  int b = i / 1088, j = i - b * 1088;
  float v;
  if (j < 512)       v = h[b * H_N + (j & 127)];
  else if (j < 1024) v = c[b * H_N + (j & 127)];
  else               v = (float)cond[b * E_N + (j - 1024)];
  out[i] = (half_t)v;
}

__global__ void k_z(const float* __restrict__ enc_out, const float* __restrict__ prior_out,
                    const float* __restrict__ eps, const half_t* __restrict__ cond,
                    float* __restrict__ out, half_t* __restrict__ dec_in) {
  int i = blockIdx.x * 256 + threadIdx.x;
  if (i >= B_N * Z_N) return;
  int b = i >> 6, j = i & 63;
  float mean = enc_out[b * 128 + j];
  float lv   = enc_out[b * 128 + 64 + j];
  float z = mean + eps[i] * __expf(0.5f * lv);
  out[i]          = z;
  out[8192 + i]   = mean;
  out[16384 + i]  = lv;
  out[24576 + i]  = prior_out[b * 128 + j];
  out[32768 + i]  = prior_out[b * 128 + 64 + j];
  dec_in[b * 128 + j]      = (half_t)z;
  dec_in[b * 128 + 64 + j] = cond[i];
}

// ------------------------------------------------------- generic WMMA GEMM
// C[M,N] = act(A[M,K] @ W[N,K]^T + bias); one 16x16 tile per wave.
// nt=1 -> non-temporal f32 stores (streaming output, never re-read).
__global__ __launch_bounds__(128) void k_gemm(const half_t* __restrict__ Am,
                                              const half_t* __restrict__ Wm,
                                              const float* __restrict__ bias,
                                              float* __restrict__ Cf, half_t* __restrict__ Ch,
                                              int M, int N, int K, int relu, int nt) {
  const int wave = threadIdx.x >> 5, lane = threadIdx.x & 31;
  const int m0 = (blockIdx.y * 4 + wave) * 16;
  const int n0 = blockIdx.x * 16;
  if (m0 >= M) return;                       // wave-uniform; EXEC stays full
  v8f acc = zero8();
  for (int k0 = 0; k0 < K; k0 += 32)
    acc = wmma16(load_a_frag(Am, K, m0, k0, lane), load_b_frag(Wm, K, n0, k0, lane), acc);
  const int col = lane & 15, rb = (lane >> 4) * 8;
#pragma unroll
  for (int j = 0; j < 8; ++j) {
    const int m = m0 + rb + j, n = n0 + col;
    float v = acc[j] + bias[n];
    if (relu) v = fmaxf(v, 0.0f);
    if (Cf) {
      if (nt) __builtin_nontemporal_store(v, &Cf[(size_t)m * N + n]);
      else    Cf[(size_t)m * N + n] = v;
    }
    if (Ch) Ch[(size_t)m * N + n] = (half_t)v;
  }
}

// --------------------------------------------------------------- LSTM args
struct CellW {
  const half_t* wih[4];
  const half_t* whh[4];
  const float*  bih[4];
  const float*  bhh[4];
};
struct EncArgs {
  CellW w;
  const half_t* tok;  // (T*B, E) f16
  const int* x;       // (T, B)
  float* h_out;       // (B, H)
  float* c_out;       // (B, H)
};
struct DecArgs {
  CellW w;
  const half_t* tok;  // (T*B, E) f16
  const float* recon; // (B, 1024)
  half_t* outs;       // (T*B, H) f16
};

#define ENC_LDS (16 * G_N * 2 + 2 * 16 * H_N * 2)      // gates + hbuf + zbuf
#define DEC_LDS (16 * G_N * 2 + 4 * 16 * H_N * 2)      // gates + 4 layer hbufs

// Encoder: 8 WGs x 16 batch rows, shared (h,c) across layers per reference.
__global__ __launch_bounds__(256) void k_encoder(EncArgs A) {
  extern __shared__ char smem[];
  half_t* gates = (half_t*)smem;                        // 16 x 512
  half_t* hbuf  = (half_t*)(smem + 16 * G_N * 2);       // 16 x 128 (masked h)
  half_t* zbuf  = hbuf + 16 * H_N;                      // 16 x 128 (layer input)
  const int tid  = threadIdx.x;
  const int wave = tid >> 5, lane = tid & 31;
  const int col  = lane & 15, rb = (lane >> 4) * 8;
  const int bl   = tid >> 4;                            // local batch row
  const int hb   = (tid & 15) * 8;                      // 8 hidden elems owned
  const int bg   = blockIdx.x * 16 + bl;
  float creg[8], hreg[8];
#pragma unroll
  for (int e = 0; e < 8; ++e) { creg[e] = 0.f; hreg[e] = 0.f; }
  for (int i = tid; i < 16 * H_N; i += 256) hbuf[i] = (half_t)0.f;
  __syncthreads();

  for (int t = 0; t < T_N; ++t) {
    for (int i = tid; i < 16 * E_N; i += 256) {
      int b = i >> 6, e = i & 63;
      zbuf[b * H_N + e] = A.tok[(size_t)t * B_N * E_N + (size_t)(blockIdx.x * 16 + b) * E_N + e];
    }
    const bool mt = (A.x[t * B_N + bg] != 0);
#pragma unroll
    for (int l = 0; l < 4; ++l) {
      __syncthreads();
      const int din = (l == 0) ? E_N : H_N;
      const half_t* wih = A.w.wih[l];
      const half_t* whh = A.w.whh[l];
      v8f acc[4];
#pragma unroll
      for (int n = 0; n < 4; ++n) acc[n] = zero8();
      for (int k0 = 0; k0 < din; k0 += 32) {
        v16h a = load_a_frag(zbuf, H_N, 0, k0, lane);
#pragma unroll
        for (int n = 0; n < 4; ++n)
          acc[n] = wmma16(a, load_b_frag(wih, din, wave * 64 + n * 16, k0, lane), acc[n]);
      }
      for (int k0 = 0; k0 < H_N; k0 += 32) {
        v16h a = load_a_frag(hbuf, H_N, 0, k0, lane);
#pragma unroll
        for (int n = 0; n < 4; ++n)
          acc[n] = wmma16(a, load_b_frag(whh, H_N, wave * 64 + n * 16, k0, lane), acc[n]);
      }
#pragma unroll
      for (int n = 0; n < 4; ++n)
#pragma unroll
        for (int j = 0; j < 8; ++j)
          gates[(rb + j) * G_N + wave * 64 + n * 16 + col] = (half_t)acc[n][j];
      __syncthreads();
      const float* bih = A.w.bih[l];
      const float* bhh = A.w.bhh[l];
#pragma unroll
      for (int e = 0; e < 8; ++e) {
        const int h = hb + e;
        float gi = (float)gates[bl * G_N + h]           + bih[h]           + bhh[h];
        float gf = (float)gates[bl * G_N + H_N + h]     + bih[H_N + h]     + bhh[H_N + h];
        float gg = (float)gates[bl * G_N + 2 * H_N + h] + bih[2 * H_N + h] + bhh[2 * H_N + h];
        float go = (float)gates[bl * G_N + 3 * H_N + h] + bih[3 * H_N + h] + bhh[3 * H_N + h];
        float c2 = sigf(gf) * creg[e] + sigf(gi) * tanhf(gg);
        float hn = sigf(go) * tanhf(c2);
        if (mt) { creg[e] = c2; hreg[e] = hn; }
        zbuf[bl * H_N + h] = (half_t)hn;        // next-layer input (unmasked)
        hbuf[bl * H_N + h] = (half_t)hreg[e];   // recurrent h (masked)
      }
    }
    __syncthreads();
  }
#pragma unroll
  for (int e = 0; e < 8; ++e) {
    A.h_out[(size_t)bg * H_N + hb + e] = hreg[e];
    A.c_out[(size_t)bg * H_N + hb + e] = creg[e];
  }
}

// Decoder: 8 WGs x 16 batch rows, per-layer states, outputs last-layer h.
__global__ __launch_bounds__(256) void k_decoder(DecArgs A) {
  extern __shared__ char smem[];
  half_t* gates = (half_t*)smem;                        // 16 x 512
  half_t* hbase = (half_t*)(smem + 16 * G_N * 2);       // 4 x 16 x 128
  const int tid  = threadIdx.x;
  const int wave = tid >> 5, lane = tid & 31;
  const int col  = lane & 15, rb = (lane >> 4) * 8;
  const int bl   = tid >> 4;
  const int hb   = (tid & 15) * 8;
  const int bg   = blockIdx.x * 16 + bl;
  float creg[4][8];
#pragma unroll
  for (int l = 0; l < 4; ++l)
#pragma unroll
    for (int e = 0; e < 8; ++e) {
      const int h = hb + e;
      hbase[l * 16 * H_N + bl * H_N + h] = (half_t)A.recon[(size_t)bg * 1024 + l * H_N + h];
      creg[l][e] = A.recon[(size_t)bg * 1024 + (4 + l) * H_N + h];
    }
  __syncthreads();

  for (int t = 0; t < T_N; ++t) {
#pragma unroll
    for (int l = 0; l < 4; ++l) {
      const int din = (l == 0) ? E_N : H_N;
      const half_t* asrc = (l == 0) ? (A.tok + (size_t)t * B_N * E_N)
                                    : (hbase + (l - 1) * 16 * H_N);
      const int lda = (l == 0) ? E_N : H_N;
      const int am0 = (l == 0) ? blockIdx.x * 16 : 0;
      half_t* hl = hbase + l * 16 * H_N;
      v8f acc[4];
#pragma unroll
      for (int n = 0; n < 4; ++n) acc[n] = zero8();
      for (int k0 = 0; k0 < din; k0 += 32) {
        v16h a = load_a_frag(asrc, lda, am0, k0, lane);
#pragma unroll
        for (int n = 0; n < 4; ++n)
          acc[n] = wmma16(a, load_b_frag(A.w.wih[l], din, wave * 64 + n * 16, k0, lane), acc[n]);
      }
      for (int k0 = 0; k0 < H_N; k0 += 32) {
        v16h a = load_a_frag(hl, H_N, 0, k0, lane);
#pragma unroll
        for (int n = 0; n < 4; ++n)
          acc[n] = wmma16(a, load_b_frag(A.w.whh[l], H_N, wave * 64 + n * 16, k0, lane), acc[n]);
      }
#pragma unroll
      for (int n = 0; n < 4; ++n)
#pragma unroll
        for (int j = 0; j < 8; ++j)
          gates[(rb + j) * G_N + wave * 64 + n * 16 + col] = (half_t)acc[n][j];
      __syncthreads();
      const float* bih = A.w.bih[l];
      const float* bhh = A.w.bhh[l];
#pragma unroll
      for (int e = 0; e < 8; ++e) {
        const int h = hb + e;
        float gi = (float)gates[bl * G_N + h]           + bih[h]           + bhh[h];
        float gf = (float)gates[bl * G_N + H_N + h]     + bih[H_N + h]     + bhh[H_N + h];
        float gg = (float)gates[bl * G_N + 2 * H_N + h] + bih[2 * H_N + h] + bhh[2 * H_N + h];
        float go = (float)gates[bl * G_N + 3 * H_N + h] + bih[3 * H_N + h] + bhh[3 * H_N + h];
        float c2 = sigf(gf) * creg[l][e] + sigf(gi) * tanhf(gg);
        float hn = sigf(go) * tanhf(c2);
        creg[l][e] = c2;
        hl[bl * H_N + h] = (half_t)hn;
        if (l == 3) A.outs[((size_t)t * B_N + bg) * H_N + h] = (half_t)hn;
      }
      __syncthreads();
    }
  }
}

// ------------------------------------------------------------------- host
extern "C" void kernel_launch(void* const* d_in, const int* in_sizes, int n_in,
                              void* d_out, int out_size, void* d_ws, size_t ws_size,
                              hipStream_t stream) {
  (void)in_sizes; (void)n_in; (void)out_size; (void)ws_size;
  const int*   xp    = (const int*)d_in[0];
  const int*   condp = (const int*)d_in[1];
  const float* epsp  = (const float*)d_in[2];

  char* wsp = (char*)d_ws;
  size_t off = 0;
  auto allocH = [&](size_t n) {
    half_t* p = (half_t*)(wsp + off);
    off += ((n * sizeof(half_t) + 255) & ~(size_t)255);
    return p;
  };
  auto allocF = [&](size_t n) {
    float* p = (float*)(wsp + off);
    off += ((n * sizeof(float) + 255) & ~(size_t)255);
    return p;
  };
  auto f2h = [&](int idx, half_t* dst, size_t n) {
    k_f2h<<<dim3((unsigned)((n + 255) / 256)), dim3(256), 0, stream>>>(
        (const float*)d_in[idx], dst, (int)n);
  };

  EncArgs ea;
  DecArgs da;
  for (int l = 0; l < 4; ++l) {
    size_t din = (l == 0) ? E_N : H_N;
    half_t* wih = allocH((size_t)G_N * din);
    half_t* whh = allocH((size_t)G_N * H_N);
    f2h(6 + 4 * l + 0, wih, (size_t)G_N * din);
    f2h(6 + 4 * l + 1, whh, (size_t)G_N * H_N);
    ea.w.wih[l] = wih; ea.w.whh[l] = whh;
    ea.w.bih[l] = (const float*)d_in[6 + 4 * l + 2];
    ea.w.bhh[l] = (const float*)d_in[6 + 4 * l + 3];
  }
  for (int l = 0; l < 4; ++l) {
    size_t din = (l == 0) ? E_N : H_N;
    half_t* wih = allocH((size_t)G_N * din);
    half_t* whh = allocH((size_t)G_N * H_N);
    f2h(22 + 4 * l + 0, wih, (size_t)G_N * din);
    f2h(22 + 4 * l + 1, whh, (size_t)G_N * H_N);
    da.w.wih[l] = wih; da.w.whh[l] = whh;
    da.w.bih[l] = (const float*)d_in[22 + 4 * l + 2];
    da.w.bhh[l] = (const float*)d_in[22 + 4 * l + 3];
  }
  half_t* wfc = allocH((size_t)V_N * H_N);
  f2h(38, wfc, (size_t)V_N * H_N);

  half_t* wce[4];
  for (int i = 0; i < 4; ++i) {
    size_t nrows = (i < 3) ? 1088 : 128;
    wce[i] = allocH(nrows * 1088);
    f2h(40 + 2 * i, wce[i], nrows * 1088);
  }
  half_t* wcp[4];
  for (int i = 0; i < 4; ++i) {
    size_t k = (i == 0) ? 64 : 128;
    wcp[i] = allocH((size_t)128 * k);
    f2h(48 + 2 * i, wcp[i], (size_t)128 * k);
  }
  half_t* wcd[4];
  for (int i = 0; i < 4; ++i) {
    size_t k = (i == 0) ? 128 : 1024;
    wcd[i] = allocH((size_t)1024 * k);
    f2h(56 + 2 * i, wcd[i], (size_t)1024 * k);
  }

  half_t* enc_tok = allocH((size_t)T_N * B_N * E_N);
  half_t* dec_tok = allocH((size_t)T_N * B_N * E_N);
  half_t* cond_h  = allocH((size_t)B_N * E_N);
  half_t* enc_in  = allocH((size_t)B_N * 1088);
  half_t* actA    = allocH((size_t)B_N * 1088);
  half_t* actB    = allocH((size_t)B_N * 1088);
  half_t* pactA   = allocH((size_t)B_N * 128);
  half_t* pactB   = allocH((size_t)B_N * 128);
  half_t* dec_in  = allocH((size_t)B_N * 128);
  half_t* outs_h  = allocH((size_t)T_N * B_N * H_N);
  float* h_sh      = allocF((size_t)B_N * H_N);
  float* c_sh      = allocF((size_t)B_N * H_N);
  float* enc_outF  = allocF((size_t)B_N * 128);
  float* prior_out = allocF((size_t)B_N * 128);
  float* recon     = allocF((size_t)B_N * 1024);

  // embeddings (row 0 zeroed per _emb)
  k_embed<<<dim3(T_N * B_N), dim3(E_N), 0, stream>>>(xp, (const float*)d_in[3], enc_tok, 0);
  k_embed<<<dim3(T_N * B_N), dim3(E_N), 0, stream>>>(xp, (const float*)d_in[4], dec_tok, 1);
  k_embed<<<dim3(B_N), dim3(E_N), 0, stream>>>(condp, (const float*)d_in[5], cond_h, 0);

  // encoder LSTM
  ea.tok = enc_tok; ea.x = xp; ea.h_out = h_sh; ea.c_out = c_sh;
  k_encoder<<<dim3(8), dim3(256), ENC_LDS, stream>>>(ea);

  // CVAE input build + MLPs
  k_build_encin<<<dim3((B_N * 1088 + 255) / 256), dim3(256), 0, stream>>>(h_sh, c_sh, cond_h, enc_in);
  auto gemm = [&](const half_t* Aa, const half_t* Ww, int bidx, float* Cf, half_t* Ch,
                  int M, int N, int K, int relu, int nt) {
    k_gemm<<<dim3(N / 16, M / 64), dim3(128), 0, stream>>>(
        Aa, Ww, (const float*)d_in[bidx], Cf, Ch, M, N, K, relu, nt);
  };
  gemm(enc_in, wce[0], 41, nullptr, actA, 128, 1088, 1088, 1, 0);
  gemm(actA,   wce[1], 43, nullptr, actB, 128, 1088, 1088, 1, 0);
  gemm(actB,   wce[2], 45, nullptr, actA, 128, 1088, 1088, 1, 0);
  gemm(actA,   wce[3], 47, enc_outF, nullptr, 128, 128, 1088, 0, 0);

  gemm(cond_h, wcp[0], 49, nullptr, pactA, 128, 128, 64, 1, 0);
  gemm(pactA,  wcp[1], 51, nullptr, pactB, 128, 128, 128, 1, 0);
  gemm(pactB,  wcp[2], 53, nullptr, pactA, 128, 128, 128, 1, 0);
  gemm(pactA,  wcp[3], 55, prior_out, nullptr, 128, 128, 128, 0, 0);

  float* out_small = (float*)d_out + (size_t)T_N * B_N * V_N;
  k_z<<<dim3(32), dim3(256), 0, stream>>>(enc_outF, prior_out, epsp, cond_h, out_small, dec_in);

  gemm(dec_in, wcd[0], 57, nullptr, actA, 128, 1024, 128, 1, 0);
  gemm(actA,   wcd[1], 59, nullptr, actB, 128, 1024, 1024, 1, 0);
  gemm(actB,   wcd[2], 61, nullptr, actA, 128, 1024, 1024, 1, 0);
  gemm(actA,   wcd[3], 63, recon, nullptr, 128, 1024, 1024, 0, 0);

  // decoder LSTM
  da.tok = dec_tok; da.recon = recon; da.outs = outs_h;
  k_decoder<<<dim3(8), dim3(256), DEC_LDS, stream>>>(da);

  // logits: (12288 x 16000) = outs @ W_fc^T + b -> d_out; non-temporal stores
  // (write-bandwidth bound: 786MB f32, never re-read, must not evict weights)
  gemm(outs_h, wfc, 39, (float*)d_out, nullptr, T_N * B_N, V_N, H_N, 0, 1);
}
